// SyncGrokMoeBlock_33724083208679
// MI455X (gfx1250) — compile-verified, matrix operand order checked by reading, not run on hardware
//
#include <hip/hip_runtime.h>
#include <hip/hip_bf16.h>
#include <math.h>

typedef __bf16 bf16;
typedef __attribute__((ext_vector_type(16))) __bf16 v16bf;
typedef __attribute__((ext_vector_type(8)))  float  v8f;

static constexpr int Bc = 4, Sc = 4096, Hc = 1024, Fc = 4096, Ec = 8, Kc = 2;
static constexpr int Tc = Bc * Sc;                       // 16384 tokens
static constexpr int FCHUNK = 512;                       // F processed in chunks (LDS budget)
static constexpr int ROWS_PER_BLK = 16;                  // WMMA M
static constexpr int MAXROWS = Tc * Kc + Ec * ROWS_PER_BLK;   // 32896 padded rows
static constexpr int MAXBLK  = MAXROWS / ROWS_PER_BLK;        // 2056 blocks

// ---------------------------------------------------------------- utilities

__global__ void zero_f32_kernel(float* __restrict__ p, size_t n) {
    size_t i = (size_t)blockIdx.x * blockDim.x + threadIdx.x;
    size_t stride = (size_t)gridDim.x * blockDim.x;
    for (; i < n; i += stride) p[i] = 0.0f;
}

__global__ void cvt_bf16_kernel(const float* __restrict__ in, bf16* __restrict__ out, size_t n) {
    size_t i = (size_t)blockIdx.x * blockDim.x + threadIdx.x;
    size_t stride = (size_t)gridDim.x * blockDim.x;
    for (; i < n; i += stride) out[i] = (bf16)in[i];
}

__global__ void init_routing_kernel(int* __restrict__ ptok, float* __restrict__ pwt,
                                    int* __restrict__ counts, int* __restrict__ cursor) {
    int i = blockIdx.x * blockDim.x + threadIdx.x;
    if (i < MAXROWS) { ptok[i] = -1; pwt[i] = 0.0f; }
    if (i < Ec) { counts[i] = 0; cursor[i] = 0; }
}

// ---------------------------------------------------------------- router
// one wave (32 lanes) per token: 8 dot products over H, softmax, top-2
__global__ __launch_bounds__(256)
void router_kernel(const float* __restrict__ x, const float* __restrict__ gw,
                   float* __restrict__ logits, int* __restrict__ sel,
                   float* __restrict__ wts) {
    const int lane = threadIdx.x & 31;
    const int t = blockIdx.x * 8 + (threadIdx.x >> 5);
    if (t >= Tc) return;
    const float* xr = x + (size_t)t * Hc;

    float acc[Ec];
#pragma unroll
    for (int e = 0; e < Ec; ++e) acc[e] = 0.0f;

    for (int h = lane; h < Hc; h += 32) {
        float xv = xr[h];
#pragma unroll
        for (int e = 0; e < Ec; ++e) acc[e] = fmaf(xv, gw[e * Hc + h], acc[e]);
    }
#pragma unroll
    for (int e = 0; e < Ec; ++e) {
        acc[e] += __shfl_xor(acc[e], 16, 32);
        acc[e] += __shfl_xor(acc[e],  8, 32);
        acc[e] += __shfl_xor(acc[e],  4, 32);
        acc[e] += __shfl_xor(acc[e],  2, 32);
        acc[e] += __shfl_xor(acc[e],  1, 32);
    }
    if (lane == 0) {
        float m = acc[0];
#pragma unroll
        for (int e = 1; e < Ec; ++e) m = fmaxf(m, acc[e]);
        float p[Ec], den = 0.0f;
#pragma unroll
        for (int e = 0; e < Ec; ++e) { p[e] = expf(acc[e] - m); den += p[e]; }
        float inv = 1.0f / den;
#pragma unroll
        for (int e = 0; e < Ec; ++e) p[e] *= inv;
        int i0 = 0;
#pragma unroll
        for (int e = 1; e < Ec; ++e) if (p[e] > p[i0]) i0 = e;
        int i1 = (i0 == 0) ? 1 : 0;
#pragma unroll
        for (int e = 0; e < Ec; ++e) if (e != i0 && p[e] > p[i1]) i1 = e;
        sel[t * 2 + 0] = i0;  wts[t * 2 + 0] = p[i0];
        sel[t * 2 + 1] = i1;  wts[t * 2 + 1] = p[i1];
#pragma unroll
        for (int e = 0; e < Ec; ++e) logits[(size_t)t * Ec + e] = acc[e];
    }
}

// ---------------------------------------------------------------- grouping
__global__ void count_kernel(const int* __restrict__ sel, int* __restrict__ counts) {
    int i = blockIdx.x * blockDim.x + threadIdx.x;
    if (i < Tc * Kc) atomicAdd(&counts[sel[i]], 1);
}

__global__ void offsets_kernel(const int* __restrict__ counts, int* __restrict__ offs,
                               int* __restrict__ blk_expert, int* __restrict__ nblk) {
    int off = 0, b = 0;
    for (int e = 0; e < Ec; ++e) {
        offs[e] = off;
        int padded = ((counts[e] + 15) >> 4) << 4;
        int nb = padded >> 4;
        for (int i = 0; i < nb; ++i) blk_expert[b++] = e;
        off += padded;
    }
    offs[Ec] = off;
    nblk[0] = b;
}

__global__ void scatter_kernel(const int* __restrict__ sel, const float* __restrict__ wts,
                               const int* __restrict__ offs, int* __restrict__ cursor,
                               int* __restrict__ ptok, float* __restrict__ pwt) {
    int i = blockIdx.x * blockDim.x + threadIdx.x;
    if (i < Tc * Kc) {
        int e = sel[i];
        int pos = atomicAdd(&cursor[e], 1);
        int idx = offs[e] + pos;
        ptok[idx] = i >> 1;     // token id
        pwt[idx]  = wts[i];     // routing weight
    }
}

// ---------------------------------------------------------------- fused expert GEMM
// block = 256 threads = 8 waves; block owns 16 gathered token rows of one expert.
__device__ inline float gelu_tanh(float g) {
    float x3 = g * g * g;
    return 0.5f * g * (1.0f + tanhf(0.7978845608028654f * (g + 0.044715f * x3)));
}

__global__ __launch_bounds__(256)
void moe_expert_kernel(const bf16* __restrict__ xb,  const bf16* __restrict__ wgb,
                       const bf16* __restrict__ wub, const bf16* __restrict__ wdb,
                       const int* __restrict__ ptok, const float* __restrict__ pwt,
                       const int* __restrict__ blk_expert, const int* __restrict__ nblk,
                       float* __restrict__ out) {
    if ((int)blockIdx.x >= nblk[0]) return;
    const int e = blk_expert[blockIdx.x];
    const int rowbase = blockIdx.x * ROWS_PER_BLK;

    __shared__ __align__(32) bf16 Xs[ROWS_PER_BLK * Hc];      // 32 KB: gathered rows
    __shared__ __align__(32) bf16 Hs[ROWS_PER_BLK * FCHUNK];  // 16 KB: gelu*up chunk
    __shared__ int   Tok[ROWS_PER_BLK];
    __shared__ float Wt[ROWS_PER_BLK];

    const int tid = threadIdx.x;
    if (tid < ROWS_PER_BLK) { Tok[tid] = ptok[rowbase + tid]; Wt[tid] = pwt[rowbase + tid]; }
    __syncthreads();

    { // stage X rows (bf16) into LDS via the CDNA5 async global->LDS path (no VGPR
      // round-trip, tracked by ASYNCcnt). Sentinel (padding) rows -> plain zero stores.
        uint4* dstv = (uint4*)Xs;
        const int vecPerRow = Hc / 8;  // 128 x 16B vectors per row
        for (int v = tid; v < ROWS_PER_BLK * vecPerRow; v += 256) {
            int m = v >> 7, c = v & 127;
            int tok = Tok[m];
            if (tok >= 0) {
                const uint4* src = ((const uint4*)(xb + (size_t)tok * Hc)) + c;
                unsigned ldsOff = (unsigned)(uintptr_t)(dstv + m * vecPerRow + c);
                asm volatile("global_load_async_to_lds_b128 %0, %1, off"
                             :: "v"(ldsOff), "v"(src)
                             : "memory");
            } else {
                uint4 z; z.x = 0u; z.y = 0u; z.z = 0u; z.w = 0u;
                dstv[m * vecPerRow + c] = z;
            }
        }
        asm volatile("s_wait_asynccnt 0" ::: "memory");
    }
    __syncthreads();

    const int lane  = tid & 31;
    const int wave  = tid >> 5;
    const int mrow  = lane & 15;          // A row / B column within tile
    const int khalf = (lane >> 4) * 16;   // K half selector (contiguous k = 16h + s, same for A and B)

    v8f zero = {};
    v8f accOut[8];
#pragma unroll
    for (int j = 0; j < 8; ++j) accOut[j] = zero;

    const bf16* wg_e = wgb + (size_t)e * Fc * Hc;
    const bf16* wu_e = wub + (size_t)e * Fc * Hc;
    const bf16* wd_e = wdb + (size_t)e * Hc * Fc;

    for (int fc = 0; fc < Fc; fc += FCHUNK) {
        // ---- phase 1: Hs = gelu(X @ Wg^T) * (X @ Wu^T) for cols [fc, fc+FCHUNK)
#pragma unroll
        for (int jt = 0; jt < FCHUNK / (16 * 8); ++jt) {  // 4 N-tiles per wave
            const int f0 = fc + (wave * (FCHUNK / (16 * 8)) + jt) * 16;
            const bf16* bg = wg_e + (size_t)(f0 + mrow) * Hc + khalf;
            const bf16* bu = wu_e + (size_t)(f0 + mrow) * Hc + khalf;
            __builtin_prefetch(bg + 16 * Hc, 0, 1);   // next tile's rows
            __builtin_prefetch(bu + 16 * Hc, 0, 1);
            v8f ag = zero, au = zero;
            for (int k = 0; k < Hc; k += 32) {
                v16bf A  = *(const v16bf*)&Xs[mrow * Hc + k + khalf];
                v16bf Bg = *(const v16bf*)(bg + k);
                v16bf Bu = *(const v16bf*)(bu + k);
                ag = __builtin_amdgcn_wmma_f32_16x16x32_bf16(false, A, false, Bg,
                                                             (short)0, ag, false, false);
                au = __builtin_amdgcn_wmma_f32_16x16x32_bf16(false, A, false, Bu,
                                                             (short)0, au, false, false);
            }
#pragma unroll
            for (int r = 0; r < 8; ++r) {
                int m = r + (lane >> 4) * 8;   // C/D layout: VGPR r -> row r (+8 for hi lanes)
                float hv = gelu_tanh(ag[r]) * au[r];
                Hs[m * FCHUNK + (f0 - fc) + mrow] = (bf16)hv;
            }
        }
        __syncthreads();

        // ---- phase 2: accOut += Hs_chunk @ Wd[:, fc:fc+FCHUNK]^T  (wave owns 128 out cols)
#pragma unroll
        for (int jt = 0; jt < 8; ++jt) {
            const int n0 = wave * 128 + jt * 16;
            const bf16* bd = wd_e + (size_t)(n0 + mrow) * Fc + fc + khalf;
            __builtin_prefetch(bd + FCHUNK, 0, 1);    // next F-chunk of same rows
            v8f a = accOut[jt];
            for (int k = 0; k < FCHUNK; k += 32) {
                v16bf A  = *(const v16bf*)&Hs[mrow * FCHUNK + k + khalf];
                v16bf Bd = *(const v16bf*)(bd + k);
                a = __builtin_amdgcn_wmma_f32_16x16x32_bf16(false, A, false, Bd,
                                                            (short)0, a, false, false);
            }
            accOut[jt] = a;
        }
        __syncthreads();  // Hs reused next chunk
    }

    // ---- scale by routing weight; scatter-add into output (K=2 experts per token)
#pragma unroll
    for (int jt = 0; jt < 8; ++jt) {
        const int n = wave * 128 + jt * 16 + mrow;
#pragma unroll
        for (int r = 0; r < 8; ++r) {
            int m = r + (lane >> 4) * 8;
            int tok = Tok[m];
            if (tok >= 0) atomicAdd(&out[(size_t)tok * Hc + n], accOut[jt][r] * Wt[m]);
        }
    }
}

// ---------------------------------------------------------------- host launcher
extern "C" void kernel_launch(void* const* d_in, const int* in_sizes, int n_in,
                              void* d_out, int out_size, void* d_ws, size_t ws_size,
                              hipStream_t stream) {
    const float* x   = (const float*)d_in[0];  // [T, H]
    const float* gw  = (const float*)d_in[1];  // [E, H]
    const float* wg  = (const float*)d_in[2];  // [E, F, H]
    const float* wu  = (const float*)d_in[3];  // [E, F, H]
    const float* wd  = (const float*)d_in[4];  // [E, H, F]

    float* out    = (float*)d_out;                       // [T, H]
    float* logits = (float*)d_out + (size_t)Tc * Hc;     // [T, E]

    // workspace carve-up (256B aligned)
    size_t off = 0;
    auto carve = [&](size_t bytes) -> void* {
        void* p = (char*)d_ws + off;
        off += (bytes + 255) & ~(size_t)255;
        return p;
    };
    bf16*  xb      = (bf16*)carve((size_t)Tc * Hc * sizeof(bf16));
    bf16*  wgb     = (bf16*)carve((size_t)Ec * Fc * Hc * sizeof(bf16));
    bf16*  wub     = (bf16*)carve((size_t)Ec * Fc * Hc * sizeof(bf16));
    bf16*  wdb     = (bf16*)carve((size_t)Ec * Hc * Fc * sizeof(bf16));
    int*   sel     = (int*)carve((size_t)Tc * Kc * sizeof(int));
    float* wts     = (float*)carve((size_t)Tc * Kc * sizeof(float));
    int*   counts  = (int*)carve(Ec * sizeof(int));
    int*   offs    = (int*)carve((Ec + 1) * sizeof(int));
    int*   cursor  = (int*)carve(Ec * sizeof(int));
    int*   nblk    = (int*)carve(sizeof(int));
    int*   blk_exp = (int*)carve(MAXBLK * sizeof(int));
    int*   ptok    = (int*)carve(MAXROWS * sizeof(int));
    float* pwt     = (float*)carve(MAXROWS * sizeof(float));

    // 1. zero output accumulation region; init routing scratch
    zero_f32_kernel<<<4096, 256, 0, stream>>>(out, (size_t)Tc * Hc);
    init_routing_kernel<<<(MAXROWS + 255) / 256, 256, 0, stream>>>(ptok, pwt, counts, cursor);

    // 2. fp32 -> bf16 conversion (weights ~fit in 192MB L2 as bf16)
    cvt_bf16_kernel<<<4096, 256, 0, stream>>>(x,  xb,  (size_t)Tc * Hc);
    cvt_bf16_kernel<<<8192, 256, 0, stream>>>(wg, wgb, (size_t)Ec * Fc * Hc);
    cvt_bf16_kernel<<<8192, 256, 0, stream>>>(wu, wub, (size_t)Ec * Fc * Hc);
    cvt_bf16_kernel<<<8192, 256, 0, stream>>>(wd, wdb, (size_t)Ec * Hc * Fc);

    // 3. router (also writes router_logits output)
    router_kernel<<<Tc / 8, 256, 0, stream>>>(x, gw, logits, sel, wts);

    // 4. group (token, expert-slot) rows by expert, 16-row aligned
    count_kernel<<<(Tc * Kc + 255) / 256, 256, 0, stream>>>(sel, counts);
    offsets_kernel<<<1, 1, 0, stream>>>(counts, offs, blk_exp, nblk);
    scatter_kernel<<<(Tc * Kc + 255) / 256, 256, 0, stream>>>(sel, wts, offs, cursor, ptok, pwt);

    // 5. fused grouped GEMM (gate/up + gelu + down) with bf16 WMMA
    moe_expert_kernel<<<MAXBLK, 256, 0, stream>>>(xb, wgb, wub, wdb, ptok, pwt,
                                                  blk_exp, nblk, out);
}
